// Int8DynActInt4WeightLinear_2130303779340
// MI455X (gfx1250) — 8-bit matrix operand order repaired and checked by reading, compiled, not run on hardware
//
#include <hip/hip_runtime.h>
#include <stdint.h>

typedef __attribute__((ext_vector_type(8))) int v8i;

#define M_TOK 8192
#define N_OUT 4096
#define K_IN  4096
#define GSIZE 256
#define NGRP  16
#define KSTEPS (K_IN / 64)

__global__ __launch_bounds__(256) void quant_kernel(
    const float* __restrict__ x,
    int8_t* __restrict__ q,
    int* __restrict__ sq,
    float* __restrict__ st,
    float* __restrict__ szp)
{
    const int t   = blockIdx.x;
    const int tid = threadIdx.x;

    const float* xr = x + (size_t)t * K_IN + tid * 16;
    float f[16];
    {
        const float4* xp = (const float4*)xr;
        #pragma unroll
        for (int i = 0; i < 4; ++i) {
            float4 v = xp[i];
            f[i*4+0] = v.x; f[i*4+1] = v.y; f[i*4+2] = v.z; f[i*4+3] = v.w;
        }
    }

    float mn = f[0], mx = f[0];
    #pragma unroll
    for (int j = 1; j < 16; ++j) { mn = fminf(mn, f[j]); mx = fmaxf(mx, f[j]); }

    __shared__ float smn[256], smx[256];
    smn[tid] = mn; smx[tid] = mx;
    __syncthreads();
    for (int s = 128; s > 0; s >>= 1) {
        if (tid < s) {
            smn[tid] = fminf(smn[tid], smn[tid + s]);
            smx[tid] = fmaxf(smx[tid], smx[tid + s]);
        }
        __syncthreads();
    }
    mn = fminf(smn[0], 0.0f);
    mx = fmaxf(smx[0], 0.0f);

    const float scale = fmaxf((mx - mn) * (1.0f / 255.0f), 1.1920928955078125e-07f);
    const float zp    = fminf(fmaxf(-128.0f - rintf(mn / scale), -128.0f), 127.0f);
    const float inv   = 1.0f / scale;

    int lsum = 0;
    uint pk[4];
    #pragma unroll
    for (int c = 0; c < 4; ++c) {
        uint p = 0;
        #pragma unroll
        for (int j = 0; j < 4; ++j) {
            float qf = fminf(fmaxf(rintf(f[c*4+j] * inv) + zp, -128.0f), 127.0f);
            int qi = (int)qf;
            lsum += qi;
            p |= (uint)(qi & 0xff) << (8 * j);
        }
        pk[c] = p;
    }
    uint4 pv; pv.x = pk[0]; pv.y = pk[1]; pv.z = pk[2]; pv.w = pk[3];
    *(uint4*)(q + (size_t)t * K_IN + tid * 16) = pv;

    __shared__ int ssum[256];
    ssum[tid] = lsum;
    __syncthreads();
    if (tid < NGRP) {
        int s = 0;
        #pragma unroll
        for (int j = 0; j < 16; ++j) s += ssum[tid * 16 + j];
        sq[(size_t)t * NGRP + tid] = s;
    }
    if (tid == 0) {
        st[t]  = scale;
        szp[t] = scale * zp;
    }
}

__global__ __launch_bounds__(256) void corr_kernel(
    const int8_t* __restrict__ w,
    const float* __restrict__ scales,
    const float* __restrict__ zeros,
    float* __restrict__ Co)
{
    const int o = blockIdx.x * 256 + threadIdx.x;
    const int8_t* wr = w + (size_t)o * K_IN;
    float c = 0.0f;
    for (int g = 0; g < NGRP; ++g) {
        int s = 0;
        const char4* wp = (const char4*)(wr + g * GSIZE);
        #pragma unroll 8
        for (int i = 0; i < GSIZE / 4; ++i) {
            char4 cc = wp[i];
            s += (int)cc.x + (int)cc.y + (int)cc.z + (int)cc.w;
        }
        c += scales[(size_t)o * NGRP + g] * ((float)s - 256.0f * zeros[(size_t)o * NGRP + g]);
    }
    Co[o] = c;
}

__global__ __launch_bounds__(256) void wmma_gemm_kernel(
    const int8_t* __restrict__ q,
    const int8_t* __restrict__ w,
    const float*  __restrict__ scales,
    const float*  __restrict__ zeros,
    const float*  __restrict__ bias,
    const int*    __restrict__ sq,
    const float*  __restrict__ st,
    const float*  __restrict__ szp,
    const float*  __restrict__ Co,
    float* __restrict__ out)
{
    __shared__ uint4 lds_a4[3][64 * 4];
    __shared__ uint4 lds_b4[3][64 * 4];
    __shared__ int   lds_sq[64 * NGRP];
    __shared__ float lds_st[64], lds_szp[64];

    const int tid  = threadIdx.x;
    const int wave = tid >> 5;
    const int lane = tid & 31;
    const int hi   = lane >> 4;
    const int ln   = lane & 15;
    const int wm   = wave & 3;
    const int wn   = wave >> 2;

    const int m0 = blockIdx.y * 64;
    const int n0 = blockIdx.x * 64;

    for (int i = tid; i < 64 * NGRP; i += 256)
        lds_sq[i] = sq[(size_t)(m0 + (i >> 4)) * NGRP + (i & 15)];
    if (tid < 64)       lds_st[tid]       = st[m0 + tid];
    else if (tid < 128) lds_szp[tid - 64] = szp[m0 + tid - 64];

    const int lrow = tid >> 2, lch = tid & 3;
    const int8_t* gA = q + (size_t)(m0 + lrow) * K_IN + lch * 16;
    const int8_t* gB = w + (size_t)(n0 + lrow) * K_IN + lch * 16;

    auto issue = [&](int buf, int step) {
        const uint la = (uint)(uintptr_t)&lds_a4[buf][tid];
        const uint lb = (uint)(uintptr_t)&lds_b4[buf][tid];
        const int8_t* pa = gA + step * 64;
        const int8_t* pb = gB + step * 64;
        asm volatile(
            "global_load_async_to_lds_b128 %0, %2, off\n\t"
            "global_load_async_to_lds_b128 %1, %3, off"
            :: "v"(la), "v"(lb), "v"(pa), "v"(pb) : "memory");
        __builtin_prefetch(pa + 64, 0, 0);
        __builtin_prefetch(pb + 64, 0, 0);
    };

    const uint* lds_a = (const uint*)lds_a4;
    const uint* lds_b = (const uint*)lds_b4;

    const int o0 = n0 + wn * 32 + ln;
    const int o1 = o0 + 16;

    float facc0[8] = {0,0,0,0,0,0,0,0};
    float facc1[8] = {0,0,0,0,0,0,0,0};

    issue(0, 0);
    issue(1, 1);
    int cur = 0;
    int nxt = 2;

    for (int g = 0; g < NGRP; ++g) {
        v8i iacc0 = {}, iacc1 = {};
        const float sg0 = scales[(size_t)o0 * NGRP + g];
        const float zg0 = zeros [(size_t)o0 * NGRP + g];
        const float sg1 = scales[(size_t)o1 * NGRP + g];
        const float zg1 = zeros [(size_t)o1 * NGRP + g];

        #pragma unroll
        for (int kk = 0; kk < 4; ++kk) {
            const int step = g * 4 + kk;

            if (step + 1 < KSTEPS)
                asm volatile("s_wait_asynccnt 0x2" ::: "memory");
            else
                asm volatile("s_wait_asynccnt 0x0" ::: "memory");
            __syncthreads();
            if (step + 2 < KSTEPS) issue(nxt, step + 2);

            const uint* Ar = lds_a + cur * (64 * 16) + (wm * 16 + ln) * 16;
            const uint* B0 = lds_b + cur * (64 * 16) + (wn * 32 + ln) * 16;
            const uint* B1 = B0 + 16 * 16;
            v8i a, b0, b1;
            #pragma unroll
            for (int j = 0; j < 8; ++j) {
                const int ad = (j >> 1) * 4 + (j & 1) + hi * 2;
                const int bd = (j >> 1) * 4 + (j & 1) + hi * 2;
                a [j] = (int)Ar[ad];
                b0[j] = (int)B0[bd];
                b1[j] = (int)B1[bd];
            }
            iacc0 = __builtin_amdgcn_wmma_i32_16x16x64_iu8(
                true, a, true, b0, iacc0, false, false);
            iacc1 = __builtin_amdgcn_wmma_i32_16x16x64_iu8(
                true, a, true, b1, iacc1, false, false);

            cur = (cur == 2) ? 0 : cur + 1;
            nxt = (nxt == 2) ? 0 : nxt + 1;
        }

        const float sz0 = sg0 * zg0, sz1 = sg1 * zg1;
        #pragma unroll
        for (int j = 0; j < 8; ++j) {
            const int rowl = wm * 16 + hi * 8 + j;
            const float sqv = (float)lds_sq[rowl * NGRP + g];
            facc0[j] += sg0 * (float)iacc0[j] - sz0 * sqv;
            facc1[j] += sg1 * (float)iacc1[j] - sz1 * sqv;
        }
    }

    const float co0 = Co[o0], bo0 = bias[o0];
    const float co1 = Co[o1], bo1 = bias[o1];
    #pragma unroll
    for (int j = 0; j < 8; ++j) {
        const int rowl = wm * 16 + hi * 8 + j;
        const float stv = lds_st[rowl], szpv = lds_szp[rowl];
        float* orow = out + (size_t)(m0 + rowl) * N_OUT;
        orow[o0] = stv * facc0[j] - szpv * co0 + bo0;
        orow[o1] = stv * facc1[j] - szpv * co1 + bo1;
    }
}

extern "C" void kernel_launch(void* const* d_in, const int* in_sizes, int n_in,
                              void* d_out, int out_size, void* d_ws, size_t ws_size,
                              hipStream_t stream) {
    (void)in_sizes; (void)n_in; (void)out_size; (void)ws_size;

    const float*  x      = (const float*)d_in[0];
    const int8_t* w      = (const int8_t*)d_in[1];
    const float*  scales = (const float*)d_in[2];
    const float*  zeros  = (const float*)d_in[3];
    const float*  bias   = (const float*)d_in[4];
    float* out = (float*)d_out;

    char* ws = (char*)d_ws;
    int8_t* q  = (int8_t*)ws;
    int*    sq = (int*)  (ws + 33554432);
    float*  st = (float*)(ws + 33554432 + 524288);
    float*  zp = (float*)(ws + 33554432 + 524288 + 32768);
    float*  Co = (float*)(ws + 33554432 + 524288 + 65536);

    quant_kernel<<<M_TOK, 256, 0, stream>>>(x, q, sq, st, zp);
    corr_kernel<<<N_OUT / 256, 256, 0, stream>>>(w, scales, zeros, Co);

    dim3 grid(N_OUT / 64, M_TOK / 64);
    wmma_gemm_kernel<<<grid, 256, 0, stream>>>(q, w, scales, zeros, bias,
                                               sq, st, zp, Co, out);
}
